// Int8Linear_6777458393778
// MI455X (gfx1250) — 8-bit matrix operand order repaired and checked by reading, compiled, not run on hardware
//
#include <hip/hip_runtime.h>
#include <hip/hip_bf16.h>

typedef __attribute__((ext_vector_type(8))) int          v8i;
typedef __attribute__((ext_vector_type(4))) int          v4i;
typedef __attribute__((ext_vector_type(4))) unsigned int v4u;
typedef __attribute__((ext_vector_type(8))) float        v8f;

#define K_DIM   4096
#define N_DIM   12288
#define M_DIM   8192
#define KBLK    32
#define PADROW  144
#define TILE_BYTES (128 * PADROW)

#if defined(__has_builtin)
#if __has_builtin(__builtin_amdgcn_tensor_load_to_lds) && __has_builtin(__builtin_amdgcn_s_wait_tensorcnt)
#define HAVE_TDM 1
#endif
#endif

__global__ __launch_bounds__(256)
void act_quant_kernel(const float* __restrict__ x,
                      signed char* __restrict__ q,
                      float* __restrict__ ascale)
{
    const int gwave = (blockIdx.x * blockDim.x + threadIdx.x) >> 5;
    const int lane  = threadIdx.x & 31;
    const int row   = gwave >> 5;
    const int kb    = gwave & 31;

    const float4 v = *(const float4*)(x + (size_t)row * K_DIM + kb * 128 + lane * 4);

    float m = fmaxf(fmaxf(fabsf(v.x), fabsf(v.y)), fmaxf(fabsf(v.z), fabsf(v.w)));
    #pragma unroll
    for (int off = 16; off > 0; off >>= 1)
        m = fmaxf(m, __shfl_xor(m, off, 32));

    const float scale = fmaxf(m * (1.0f / 127.0f), 1e-8f);
    const float inv   = 127.0f / fmaxf(m, 127.0f * 1e-8f);

    int a0 = (int)rintf(v.x * inv);
    int a1 = (int)rintf(v.y * inv);
    int a2 = (int)rintf(v.z * inv);
    int a3 = (int)rintf(v.w * inv);
    a0 = min(127, max(-127, a0));
    a1 = min(127, max(-127, a1));
    a2 = min(127, max(-127, a2));
    a3 = min(127, max(-127, a3));

    const int packed = (a0 & 0xFF) | ((a1 & 0xFF) << 8) | ((a2 & 0xFF) << 16) | (a3 << 24);
    ((int*)q)[(size_t)row * (K_DIM / 4) + kb * 32 + lane] = packed;

    if (lane == 0)
        ascale[row * KBLK + kb] = scale;
}

__global__ __launch_bounds__(256)
void w_pack_kernel(const int* __restrict__ w, signed char* __restrict__ wq, int n4)
{
    const int i = blockIdx.x * blockDim.x + threadIdx.x;
    if (i < n4) {
        const int4 v = ((const int4*)w)[i];
        ((int*)wq)[i] = (v.x & 0xFF) | ((v.y & 0xFF) << 8) | ((v.z & 0xFF) << 16) | (v.w << 24);
    }
}

#ifdef HAVE_TDM
__device__ __forceinline__ void tdm_load_tile(unsigned ldsOff,
                                              unsigned long long gaddr,
                                              unsigned tensorRows)
{
    v4u g0;
    g0[0] = 1u;
    g0[1] = ldsOff;
    g0[2] = (unsigned)(gaddr & 0xFFFFFFFFull);
    g0[3] = (unsigned)((gaddr >> 32) & 0x1FFFFFFu) | (2u << 30);

    v8i g1;
    g1[0] = (int)((1u << 20) | (4u << 22) | (3u << 25));
    g1[1] = (int)((4096u & 0xFFFFu) << 16);
    g1[2] = (int)((4096u >> 16) | ((tensorRows & 0xFFFFu) << 16));
    g1[3] = (int)((tensorRows >> 16) | (128u << 16));
    g1[4] = (int)128u;
    g1[5] = (int)4096;
    g1[6] = 0;
    g1[7] = 0;

    v4i z4; z4[0] = 0; z4[1] = 0; z4[2] = 0; z4[3] = 0;
    v8i z8;
    #pragma unroll
    for (int i = 0; i < 8; ++i) z8[i] = 0;
    __builtin_amdgcn_tensor_load_to_lds(g0, g1, z4, z4, z8, 0);
}
#endif

__global__ __launch_bounds__(256, 2)
void gemm_i8_kernel(const signed char* __restrict__ qa,
                    const float* __restrict__ ascale,
                    const signed char* __restrict__ qw,
                    const float* __restrict__ wscale,
                    const float* __restrict__ bias,
                    float* __restrict__ out)
{
    __shared__ __align__(16) signed char smem[4 * TILE_BYTES];

    const int tid  = threadIdx.x;
    const int lane = tid & 31;
    const int wave = tid >> 5;
    const int wm   = wave >> 1;
    const int wn   = wave & 1;

    const int rowbase = blockIdx.y * 128;
    const int nbase   = blockIdx.x * 128;

    const int h = lane >> 4;
    const int r = lane & 15;

    v8f acc[2][4];
    #pragma unroll
    for (int ms = 0; ms < 2; ++ms)
        #pragma unroll
        for (int ns = 0; ns < 4; ++ns)
            #pragma unroll
            for (int v = 0; v < 8; ++v)
                acc[ms][ns][v] = 0.0f;

    const unsigned ldsBase = (unsigned)(uintptr_t)&smem[0];
    const unsigned long long gA = (unsigned long long)(uintptr_t)(qa + (size_t)rowbase * K_DIM);
    const unsigned long long gB = (unsigned long long)(uintptr_t)(qw + (size_t)nbase   * K_DIM);

#ifdef HAVE_TDM
    if (wave == 0) {
        tdm_load_tile(ldsBase + 0 * TILE_BYTES, gA, (unsigned)M_DIM);
        tdm_load_tile(ldsBase + 2 * TILE_BYTES, gB, (unsigned)N_DIM);
    }
#else
    {
        #pragma unroll
        for (int it = 0; it < 4; ++it) {
            const int idx = it * 256 + tid;
            const int row = idx >> 3;
            const int col = (idx & 7) * 16;
            *(int4*)&smem[0 * TILE_BYTES + row * PADROW + col] =
                *(const int4*)(qa + (size_t)(rowbase + row) * K_DIM + col);
            *(int4*)&smem[2 * TILE_BYTES + row * PADROW + col] =
                *(const int4*)(qw + (size_t)(nbase + row) * K_DIM + col);
        }
    }
#endif

    const float* asP = ascale + (size_t)(rowbase + wm * 32 + 8 * h) * KBLK;
    const float* wsP = wscale + (size_t)blockIdx.x * KBLK;

    for (int kb = 0; kb < KBLK; ++kb) {
        const int buf = kb & 1;

#ifdef HAVE_TDM
        if (kb + 1 < KBLK) {
            if (wave == 0) {
                const unsigned nb = buf ^ 1;
                tdm_load_tile(ldsBase + nb * TILE_BYTES,       gA + (kb + 1) * 128, (unsigned)M_DIM);
                tdm_load_tile(ldsBase + (2 + nb) * TILE_BYTES, gB + (kb + 1) * 128, (unsigned)N_DIM);
                __builtin_amdgcn_s_wait_tensorcnt(2);
            }
        } else {
            if (wave == 0)
                __builtin_amdgcn_s_wait_tensorcnt(0);
        }
        __syncthreads();
#else
        __syncthreads();
        if (kb + 1 < KBLK) {
            const int nb = buf ^ 1;
            #pragma unroll
            for (int it = 0; it < 4; ++it) {
                const int idx = it * 256 + tid;
                const int row = idx >> 3;
                const int col = (idx & 7) * 16;
                *(int4*)&smem[nb * TILE_BYTES + row * PADROW + col] =
                    *(const int4*)(qa + (size_t)(rowbase + row) * K_DIM + (kb + 1) * 128 + col);
                *(int4*)&smem[(2 + nb) * TILE_BYTES + row * PADROW + col] =
                    *(const int4*)(qw + (size_t)(nbase + row) * K_DIM + (kb + 1) * 128 + col);
            }
        }
#endif

        const signed char* As = smem + buf * TILE_BYTES;
        const signed char* Bs = smem + (2 + buf) * TILE_BYTES;

        v8i af[2][2];
        #pragma unroll
        for (int ms = 0; ms < 2; ++ms) {
            const int arow = (wm * 32 + ms * 16 + r) * PADROW;
            #pragma unroll
            for (int kk = 0; kk < 2; ++kk) {
                const int base = arow + kk * 64 + h * 16;
                #pragma unroll
                for (int g = 0; g < 4; ++g) {
                    const int2 d = *(const int2*)(As + base + g * 8 + (g >> 1) * 16);
                    af[ms][kk][2 * g]     = d.x;
                    af[ms][kk][2 * g + 1] = d.y;
                }
            }
        }

        float as[2][8];
        #pragma unroll
        for (int ms = 0; ms < 2; ++ms)
            #pragma unroll
            for (int v = 0; v < 8; ++v)
                as[ms][v] = asP[(ms * 16 + v) * KBLK + kb];
        const float ws = wsP[kb];

        #pragma unroll
        for (int ns = 0; ns < 4; ++ns) {
            const int brow = (wn * 64 + ns * 16 + r) * PADROW;
            v8i bf0, bf1;
            {
                const int4 b0 = *(const int4*)(Bs + brow + h * 16);
                const int4 b1 = *(const int4*)(Bs + brow + h * 16 + 32);
                bf0[0] = b0.x; bf0[1] = b0.y; bf0[2] = b0.z; bf0[3] = b0.w;
                bf0[4] = b1.x; bf0[5] = b1.y; bf0[6] = b1.z; bf0[7] = b1.w;
                const int4 b2 = *(const int4*)(Bs + brow + 64 + h * 16);
                const int4 b3 = *(const int4*)(Bs + brow + 64 + h * 16 + 32);
                bf1[0] = b2.x; bf1[1] = b2.y; bf1[2] = b2.z; bf1[3] = b2.w;
                bf1[4] = b3.x; bf1[5] = b3.y; bf1[6] = b3.z; bf1[7] = b3.w;
            }
            #pragma unroll
            for (int ms = 0; ms < 2; ++ms) {
                v8i ci;
                #pragma unroll
                for (int v = 0; v < 8; ++v) ci[v] = 0;
                ci = __builtin_amdgcn_wmma_i32_16x16x64_iu8(true, af[ms][0], true, bf0, ci,
                                                            false, false);
                ci = __builtin_amdgcn_wmma_i32_16x16x64_iu8(true, af[ms][1], true, bf1, ci,
                                                            false, false);
                #pragma unroll
                for (int v = 0; v < 8; ++v)
                    acc[ms][ns][v] += (float)ci[v] * (as[ms][v] * ws);
            }
        }

        __syncthreads();
    }

    #pragma unroll
    for (int ms = 0; ms < 2; ++ms) {
        #pragma unroll
        for (int ns = 0; ns < 4; ++ns) {
            const int ncol = nbase + wn * 64 + ns * 16 + r;
            const float b  = bias[ncol];
            #pragma unroll
            for (int v = 0; v < 8; ++v) {
                const int row = rowbase + wm * 32 + ms * 16 + v + 8 * h;
                out[(size_t)row * N_DIM + ncol] = acc[ms][ns][v] + b;
            }
        }
    }
}

extern "C" void kernel_launch(void* const* d_in, const int* in_sizes, int n_in,
                              void* d_out, int out_size, void* d_ws, size_t ws_size,
                              hipStream_t stream)
{
    const float* x      = (const float*)d_in[0];
    const int*   w      = (const int*)d_in[1];
    const float* wscale = (const float*)d_in[2];
    const float* bias   = (const float*)d_in[3];
    float*       out    = (float*)d_out;

    signed char* qa     = (signed char*)d_ws;
    float*       ascale = (float*)((char*)d_ws + (size_t)M_DIM * K_DIM);
    signed char* qw     = (signed char*)((char*)d_ws + (size_t)M_DIM * K_DIM
                                         + (size_t)M_DIM * KBLK * sizeof(float));

    act_quant_kernel<<<M_DIM * KBLK / 8, 256, 0, stream>>>(x, qa, ascale);

    const int n4 = (N_DIM * K_DIM) / 4;
    w_pack_kernel<<<(n4 + 255) / 256, 256, 0, stream>>>(w, qw, n4);

    dim3 grid(N_DIM / 128, M_DIM / 128);
    gemm_i8_kernel<<<grid, 256, 0, stream>>>(qa, ascale, qw, wscale, bias, out);
}
